// MultiHeadCausalSelfAttention_38122129719447
// MI455X (gfx1250) — matrix operand order repaired and checked by reading, not compiled, not run on hardware
//
#include <hip/hip_runtime.h>

typedef __attribute__((ext_vector_type(16))) _Float16 v16h;
typedef __attribute__((ext_vector_type(8)))  float    v8f;
typedef __attribute__((ext_vector_type(4)))  unsigned int v4u;
typedef __attribute__((ext_vector_type(8)))  int      v8i;
typedef __attribute__((ext_vector_type(4)))  int      v4i;

#define WMMA_F16(a, b, c) \
  __builtin_amdgcn_wmma_f32_16x16x32_f16(false, (a), false, (b), (short)0, (c), false, false)

constexpr int Bn = 4, Tn = 2048, Dn = 1024, Hn = 16, DKn = 64;
constexpr int BT = Bn * Tn;                 // 8192 rows total
constexpr float NEG_BIG = -1e30f;

// ---------------------------------------------------------------------------
// TDM: 2D tile load global->LDS (CDNA5 tensor_load_to_lds, ISA ch.8 D# layout)
//   group0: [31:0]=count(=1), [63:32]=lds_addr, [120:64]=global tile addr,
//           [127:126]=type(2)
//   group1: w0 = data_size(2B)<<16 | pad fields
//           tensor_dim0 bits[79:48], tensor_dim1 bits[111:80],
//           tile_dim0 bits[127:112], tile_dim1 bits[143:128],
//           tensor_dim0_stride bits[207:160]
//   groups 2/3 zero => 2D tile (tile_dim2/3 = 0 = unused)
// ---------------------------------------------------------------------------
__device__ __forceinline__ void tdm_load_tile_2d(
    const void* gptr, const void* lptr,
    unsigned tile0, unsigned tile1,          // tile dims (elements)
    unsigned tensor0, unsigned tensor1,      // tensor dims (elements)
    unsigned long long stride0,              // dim0 stride (elements)
    unsigned padIval, unsigned padAmt)       // LDS pad codes (ISA encodings)
{
  unsigned long long ga = (unsigned long long)gptr;
  unsigned lds = (unsigned)(unsigned long long)lptr;   // flat LDS addr[31:0] = LDS offset
  v4u g0 = { 1u, lds, (unsigned)ga,
             (unsigned)((ga >> 32) & 0x01ffffffu) | 0x80000000u };
  unsigned w0 = (1u << 16)                              // data_size = 2 bytes
              | (1u << 20)                              // pad_enable
              | (padIval << 22) | (padAmt << 25);
  unsigned w1 = (tensor0 & 0xffffu) << 16;
  unsigned w2 = (tensor0 >> 16) | ((tensor1 & 0xffffu) << 16);
  unsigned w3 = (tensor1 >> 16) | ((tile0 & 0xffffu) << 16);
  unsigned w4 = (tile1 & 0xffffu);
  unsigned w5 = (unsigned)(stride0 & 0xffffffffu);
  unsigned w6 = (unsigned)((stride0 >> 32) & 0xffffu);
  v8i g1 = {(int)w0, (int)w1, (int)w2, (int)w3, (int)w4, (int)w5, (int)w6, 0};
  v4i gz = {0, 0, 0, 0};
#if defined(__clang_major__) && (__clang_major__ >= 23)
  v8i gz8 = {0, 0, 0, 0, 0, 0, 0, 0};
  __builtin_amdgcn_tensor_load_to_lds(g0, g1, gz, gz, gz8, 0);
#else
  __builtin_amdgcn_tensor_load_to_lds(g0, g1, gz, gz, 0);
#endif
}

// ---------------------------------------------------------------------------
// f32 -> f16 conversion (vectorized, 4 elems/thread)
// ---------------------------------------------------------------------------
__global__ void cvt_f32_to_f16_kernel(const float* __restrict__ src,
                                      _Float16* __restrict__ dst, int n4) {
  int i = blockIdx.x * blockDim.x + threadIdx.x;
  if (i < n4) {
    float4 v = reinterpret_cast<const float4*>(src)[i];
    _Float16 h[4] = {(_Float16)v.x, (_Float16)v.y, (_Float16)v.z, (_Float16)v.w};
    reinterpret_cast<uint2*>(dst)[i] = *reinterpret_cast<uint2*>(h);
  }
}

// ---------------------------------------------------------------------------
// QKV projection: qkv[row][e] = sum_d x[row][d] * Wqkv[e][d]
// One wave computes a 16x64 output tile (4 WMMA accumulators).
// Q scaled by 1/sqrt(DK). Q,K stored [b][h][t][dk]; V stored transposed
// [b][h][dk][t] so attention's P@V B-fragments are contiguous.
// ---------------------------------------------------------------------------
__global__ __launch_bounds__(32) void qkv_gemm_kernel(
    const _Float16* __restrict__ xh, const _Float16* __restrict__ wh,
    _Float16* __restrict__ Qd, _Float16* __restrict__ Kd,
    _Float16* __restrict__ Vtd) {
  const int lane = threadIdx.x;
  const int m    = lane & 15;     // A row / B col / D col
  const int hi   = lane >> 4;
  const int kb8  = hi * 8;        // unused (A fragment also uses k0)
  const int k0   = hi * 16;       // B-fragment K base

  const int row0 = blockIdx.x * 16;   // over B*T
  const int e0   = blockIdx.y * 64;   // over 3*D

  __shared__ _Float16 ot[16][72];     // transpose staging (144B pitch)

  const v8f vzero = {0.f, 0.f, 0.f, 0.f, 0.f, 0.f, 0.f, 0.f};
  v8f acc[4];
  acc[0] = acc[1] = acc[2] = acc[3] = vzero;

  const _Float16* aRow = xh + (size_t)(row0 + m) * Dn;

  for (int kk = 0; kk < Dn; kk += 32) {
    v16h a;
    ((uint4*)&a)[0] = *(const uint4*)(aRow + kk + k0);
    ((uint4*)&a)[1] = *(const uint4*)(aRow + kk + k0 + 8);
#pragma unroll
    for (int t = 0; t < 4; ++t) {
      v16h bf;  // B[k0+h][n] = W[e0+t*16+n][kk+k0+h] -> contiguous 32B
      const _Float16* bp = wh + (size_t)(e0 + t * 16 + m) * Dn + kk + k0;
      ((uint4*)&bf)[0] = ((const uint4*)bp)[0];
      ((uint4*)&bf)[1] = ((const uint4*)bp)[1];
      acc[t] = WMMA_F16(a, bf, acc[t]);
    }
  }

  const int b    = row0 >> 11;        // / Tn
  const int t0   = row0 & (Tn - 1);
  const int sec  = e0 >> 10;          // 0=q 1=k 2=v
  const int head = (e0 & (Dn - 1)) >> 6;
  const size_t bh = (size_t)(b * Hn + head);

  if (sec == 2) {
    // V transposed store: lane holds column dk=t*16+m, rows t0+8*hi+r (contiguous in t)
#pragma unroll
    for (int t = 0; t < 4; ++t) {
      _Float16 pack[8];
#pragma unroll
      for (int r = 0; r < 8; ++r) pack[r] = (_Float16)acc[t][r];
      _Float16* dst = Vtd + (bh * DKn + t * 16 + m) * Tn + t0 + 8 * hi;
      *(uint4*)dst = *(const uint4*)pack;
    }
  } else {
    const float scl = (sec == 0) ? 0.125f : 1.0f;  // 1/sqrt(64) folded into Q
#pragma unroll
    for (int t = 0; t < 4; ++t)
#pragma unroll
      for (int r = 0; r < 8; ++r)
        ot[r + 8 * hi][t * 16 + m] = (_Float16)(acc[t][r] * scl);
    __syncthreads();
    _Float16* basep = (sec == 0) ? Qd : Kd;
    const int rr = lane >> 1, cc = (lane & 1) * 32;
    _Float16* dst = basep + (bh * Tn + t0 + rr) * DKn + cc;
#pragma unroll
    for (int c = 0; c < 4; ++c)
      ((uint4*)dst)[c] = *(const uint4*)(&ot[rr][cc + c * 8]);
  }
}

// ---------------------------------------------------------------------------
// Flash attention: one wave per (b,h, 16-query block); key blocks of 32.
// K/V tiles staged into double-buffered LDS by the Tensor Data Mover
// (tensor_load_to_lds, TENSORcnt), overlapped with WMMA compute.
// S = Q @ K^T (2 N-tiles x 2 K-chained WMMAs), online softmax with
// cross-lane reductions matching the C/D layout, then O += P @ V (4 WMMAs).
// ---------------------------------------------------------------------------
__global__ __launch_bounds__(32) void flash_attn_kernel(
    const _Float16* __restrict__ Qd, const _Float16* __restrict__ Kd,
    const _Float16* __restrict__ Vtd, _Float16* __restrict__ Ah) {
  const int lane = threadIdx.x;
  const int m    = lane & 15;
  const int hi   = lane >> 4;
  const int kb8  = hi * 8;
  const int k0   = hi * 16;

  const int iq = blockIdx.x;          // T/16
  const int bh = blockIdx.y;          // B*H
  const int b  = bh >> 4, h = bh & 15;
  const int q0 = iq * 16;

  __shared__ _Float16 Kl[2][32][72];  // K block [key][dk], 144B pitch (TDM-padded)
  __shared__ _Float16 Vl[2][64][40];  // V block [dk][key],  80B pitch (TDM-padded)
  __shared__ _Float16 Pl[16][40];     // P tile [q][key]
  __shared__ _Float16 Ol[16][72];     // output transpose staging

  const _Float16* Qb = Qd + ((size_t)bh * Tn + q0) * DKn;
  const _Float16* Kb = Kd + (size_t)bh * Tn * DKn;
  const _Float16* Vb = Vtd + (size_t)bh * DKn * Tn;

  // Q A-fragments (dk 0..31 and 32..63), fixed for the whole loop
  v16h aq0, aq1;
  {
    const _Float16* qp = Qb + m * DKn;
    ((uint4*)&aq0)[0] = *(const uint4*)(qp + k0);
    ((uint4*)&aq0)[1] = *(const uint4*)(qp + k0 + 8);
    ((uint4*)&aq1)[0] = *(const uint4*)(qp + 32 + k0);
    ((uint4*)&aq1)[1] = *(const uint4*)(qp + 32 + k0 + 8);
  }

  const v8f vzero = {0.f, 0.f, 0.f, 0.f, 0.f, 0.f, 0.f, 0.f};
  v8f o[4];
  o[0] = o[1] = o[2] = o[3] = vzero;
  float rmax[8], rsum[8];
#pragma unroll
  for (int r = 0; r < 8; ++r) { rmax[r] = NEG_BIG; rsum[r] = 0.f; }

  const int nkb = (q0 + 47) >> 5;     // number of 32-wide key blocks

  // prologue: TDM-stage block 0 into buffer 0
  // K tile: 32 rows x 64 halves (128B rows -> pad 4 dw every 32 dw: codes 4/3)
  tdm_load_tile_2d(Kb, &Kl[0][0][0], 64u, 32u, 64u, 32u, 64ull, 4u, 3u);
  // V tile: 64 rows x 32 halves ( 64B rows -> pad 4 dw every 16 dw: codes 3/3)
  tdm_load_tile_2d(Vb, &Vl[0][0][0], 32u, 64u, 32u, 64u, (unsigned long long)Tn, 3u, 3u);

  for (int ib = 0; ib < nkb; ++ib) {
    const int kb  = ib * 32;
    const int buf = ib & 1;

    if (ib + 1 < nkb) {
      const int kn = kb + 32;
      tdm_load_tile_2d(Kb + (size_t)kn * DKn, &Kl[buf ^ 1][0][0],
                       64u, 32u, 64u, 32u, 64ull, 4u, 3u);
      tdm_load_tile_2d(Vb + kn, &Vl[buf ^ 1][0][0],
                       32u, 64u, 32u, 64u, (unsigned long long)Tn, 3u, 3u);
      __builtin_amdgcn_s_wait_tensorcnt(2);  // current buffer's 2 loads done
    } else {
      __builtin_amdgcn_s_wait_tensorcnt(0);
    }
    __syncthreads();

    // S tiles: keys [kb, kb+16) and [kb+16, kb+32)
    v8f s[2];
#pragma unroll
    for (int t = 0; t < 2; ++t) {
      v16h b0, b1;  // B[k0+hh][n] = K[key = t*16+n][dk = (chunk)+k0+hh]
      const _Float16* kr = &Kl[buf][t * 16 + m][0];
      ((uint4*)&b0)[0] = *(const uint4*)(kr + k0);
      ((uint4*)&b0)[1] = *(const uint4*)(kr + k0 + 8);
      ((uint4*)&b1)[0] = *(const uint4*)(kr + 32 + k0);
      ((uint4*)&b1)[1] = *(const uint4*)(kr + 32 + k0 + 8);
      v8f c = WMMA_F16(aq0, b0, vzero);
      c = WMMA_F16(aq1, b1, c);
      s[t] = c;
    }

    // causal mask (only the diagonal block(s) trigger this)
    if (kb + 31 > q0) {
#pragma unroll
      for (int t = 0; t < 2; ++t)
#pragma unroll
        for (int r = 0; r < 8; ++r)
          if (kb + t * 16 + m > q0 + r + 8 * hi) s[t][r] = NEG_BIG;
    }

    // block row-max across the 16 lanes of each half-wave
    float bm[8];
#pragma unroll
    for (int r = 0; r < 8; ++r) bm[r] = fmaxf(s[0][r], s[1][r]);
    for (int off = 1; off < 16; off <<= 1) {
#pragma unroll
      for (int r = 0; r < 8; ++r)
        bm[r] = fmaxf(bm[r], __shfl_xor(bm[r], off, 32));
    }

    // online softmax update
    float ps[8];
#pragma unroll
    for (int r = 0; r < 8; ++r) {
      float mn = fmaxf(rmax[r], bm[r]);
      float sc = __expf(rmax[r] - mn);
      rmax[r] = mn;
      rsum[r] *= sc;
      o[0][r] *= sc; o[1][r] *= sc; o[2][r] *= sc; o[3][r] *= sc;
      float p0 = __expf(s[0][r] - mn);
      float p1 = __expf(s[1][r] - mn);
      s[0][r] = p0; s[1][r] = p1;
      ps[r] = p0 + p1;
    }
    for (int off = 1; off < 16; off <<= 1) {
#pragma unroll
      for (int r = 0; r < 8; ++r) ps[r] += __shfl_xor(ps[r], off, 32);
    }
#pragma unroll
    for (int r = 0; r < 8; ++r) rsum[r] += ps[r];

    // reshape P: D-tile layout -> A-fragment layout via LDS
#pragma unroll
    for (int t = 0; t < 2; ++t)
#pragma unroll
      for (int r = 0; r < 8; ++r)
        Pl[r + 8 * hi][t * 16 + m] = (_Float16)s[t][r];
    __syncthreads();

    v16h pa;
    {
      const _Float16* pp = &Pl[m][0];
      ((uint4*)&pa)[0] = *(const uint4*)(pp + k0);
      ((uint4*)&pa)[1] = *(const uint4*)(pp + k0 + 8);
    }

    // O += P @ V : B[k0+hh][n] = Vl[dk = t*16+n][key = k0+hh] (contiguous)
#pragma unroll
    for (int t = 0; t < 4; ++t) {
      v16h bv;
      const _Float16* vr = &Vl[buf][t * 16 + m][0];
      ((uint4*)&bv)[0] = *(const uint4*)(vr + k0);
      ((uint4*)&bv)[1] = *(const uint4*)(vr + k0 + 8);
      o[t] = WMMA_F16(pa, bv, o[t]);
    }
    __syncthreads();
  }

  // finalize: divide by rowsum, write [b][t][h*64+dk] f16 via LDS transpose
#pragma unroll
  for (int r = 0; r < 8; ++r) {
    float inv = 1.0f / rsum[r];
#pragma unroll
    for (int t = 0; t < 4; ++t)
      Ol[r + 8 * hi][t * 16 + m] = (_Float16)(o[t][r] * inv);
  }
  __syncthreads();
  const int rr = lane >> 1, cc = (lane & 1) * 32;
  _Float16* dst = Ah + ((size_t)b * Tn + q0 + rr) * Dn + h * DKn + cc;
#pragma unroll
  for (int c = 0; c < 4; ++c)
    ((uint4*)dst)[c] = *(const uint4*)(&Ol[rr][cc + c * 8]);
}

// ---------------------------------------------------------------------------
// Output projection: out[row][e] = sum_d Ah[row][d] * Wo[e][d]  (f32 out)
// ---------------------------------------------------------------------------
__global__ __launch_bounds__(32) void out_proj_kernel(
    const _Float16* __restrict__ Ah, const _Float16* __restrict__ wh,
    float* __restrict__ out) {
  const int lane = threadIdx.x;
  const int m    = lane & 15;
  const int hi   = lane >> 4;
  const int kb8  = hi * 8;
  const int k0   = hi * 16;

  const int row0 = blockIdx.x * 16;
  const int e0   = blockIdx.y * 64;

  __shared__ float os[16][68];   // 272B pitch (17 x 16B)

  const v8f vzero = {0.f, 0.f, 0.f, 0.f, 0.f, 0.f, 0.f, 0.f};
  v8f acc[4];
  acc[0] = acc[1] = acc[2] = acc[3] = vzero;

  const _Float16* aRow = Ah + (size_t)(row0 + m) * Dn;

  for (int kk = 0; kk < Dn; kk += 32) {
    v16h a;
    ((uint4*)&a)[0] = *(const uint4*)(aRow + kk + k0);
    ((uint4*)&a)[1] = *(const uint4*)(aRow + kk + k0 + 8);
#pragma unroll
    for (int t = 0; t < 4; ++t) {
      v16h bf;
      const _Float16* bp = wh + (size_t)(e0 + t * 16 + m) * Dn + kk + k0;
      ((uint4*)&bf)[0] = ((const uint4*)bp)[0];
      ((uint4*)&bf)[1] = ((const uint4*)bp)[1];
      acc[t] = WMMA_F16(a, bf, acc[t]);
    }
  }

#pragma unroll
  for (int t = 0; t < 4; ++t)
#pragma unroll
    for (int r = 0; r < 8; ++r)
      os[r + 8 * hi][t * 16 + m] = acc[t][r];
  __syncthreads();
  const int rr = lane >> 1, cc = (lane & 1) * 32;
  float* dst = out + (size_t)(row0 + rr) * Dn + e0 + cc;
#pragma unroll
  for (int c = 0; c < 8; ++c)
    ((uint4*)dst)[c] = *(const uint4*)(&os[rr][cc + c * 4]);
}

// ---------------------------------------------------------------------------
// launch
// ---------------------------------------------------------------------------
extern "C" void kernel_launch(void* const* d_in, const int* in_sizes, int n_in,
                              void* d_out, int out_size, void* d_ws, size_t ws_size,
                              hipStream_t stream) {
  const float* x    = (const float*)d_in[0];
  const float* Wqkv = (const float*)d_in[1];
  const float* Wo   = (const float*)d_in[2];
  float* out        = (float*)d_out;

  const size_t xN    = (size_t)Bn * Tn * Dn;   // 8,388,608
  const size_t wqkvN = (size_t)3 * Dn * Dn;    // 3,145,728
  const size_t woN   = (size_t)Dn * Dn;        // 1,048,576

  _Float16* base = (_Float16*)d_ws;
  _Float16* xh  = base;
  _Float16* wqh = xh + xN;
  _Float16* woh = wqh + wqkvN;
  _Float16* Qd  = woh + woN;
  _Float16* Kd  = Qd + xN;      // [b][h][t][dk]
  _Float16* Vtd = Kd + xN;      // [b][h][dk][t]
  _Float16* Ah  = Vtd + xN;     // [b][t][d]
  // total: 46,137,344 halves = 88 MB of workspace

  cvt_f32_to_f16_kernel<<<(int)(xN / 4 / 256), 256, 0, stream>>>(x, xh, (int)(xN / 4));
  cvt_f32_to_f16_kernel<<<(int)(wqkvN / 4 / 256), 256, 0, stream>>>(Wqkv, wqh, (int)(wqkvN / 4));
  cvt_f32_to_f16_kernel<<<(int)(woN / 4 / 256), 256, 0, stream>>>(Wo, woh, (int)(woN / 4));

  qkv_gemm_kernel<<<dim3(BT / 16, (3 * Dn) / 64), 32, 0, stream>>>(xh, wqh, Qd, Kd, Vtd);
  flash_attn_kernel<<<dim3(Tn / 16, Bn * Hn), 32, 0, stream>>>(Qd, Kd, Vtd, Ah);
  out_proj_kernel<<<dim3(BT / 16, Dn / 64), 32, 0, stream>>>(Ah, woh, out);
}
